// GATLayer_14078902796419
// MI455X (gfx1250) — matrix operand order repaired and checked by reading, not compiled, not run on hardware
//
#include <hip/hip_runtime.h>

typedef __attribute__((ext_vector_type(16))) __bf16 v16bf;
typedef __attribute__((ext_vector_type(8)))  __bf16 v8bf;
typedef __attribute__((ext_vector_type(8)))  float  v8f;

#define HD 4
#define CH 256
#define HC 1024          // HD*CH
#define NEG_SLOPE 0.2f

// ---------------- utility kernels ----------------

__global__ void fill_u32(unsigned* __restrict__ p, unsigned v, long long n) {
  long long i = (long long)blockIdx.x * blockDim.x + threadIdx.x;
  if (i < n) p[i] = v;
}

// x (fp32 [Nv,K]) -> bf16 padded [Mp,K] (pad rows zero)
__global__ void cvt_pad(const float* __restrict__ x, __bf16* __restrict__ xb,
                        int Nv, int K, long long total) {
  long long i = (long long)blockIdx.x * blockDim.x + threadIdx.x;
  if (i >= total) return;
  int m = (int)(i / K);
  int k = (int)(i - (long long)m * K);
  xb[i] = (m < Nv) ? (__bf16)x[(size_t)m * K + k] : (__bf16)0.f;
}

// W (fp32 row-major [K,Nw]) -> Wt (bf16 row-major [Nw,K])
__global__ void cvt_transpose(const float* __restrict__ W, __bf16* __restrict__ Wt,
                              int K, int Nw) {
  long long i = (long long)blockIdx.x * blockDim.x + threadIdx.x;
  long long total = (long long)K * Nw;
  if (i >= total) return;
  int k = (int)(i / Nw);
  int n = (int)(i - (long long)k * Nw);
  Wt[(size_t)n * K + k] = (__bf16)W[i];
}

// ---------------- WMMA GEMM: C[M,Nw] = A[M,K](bf16) * Bt[Nw,K]^T ----------------
// One wave computes a 16x64 tile (4 x v_wmma_f32_16x16x32_bf16 per k-step).
__global__ void __launch_bounds__(256) gemm_wmma(
    const __bf16* __restrict__ A, const __bf16* __restrict__ Bt,
    float* __restrict__ C, const float* __restrict__ bias,
    int K, int Nw, int Mtiles, int Mvalid, int ldc) {
  int nT = Nw >> 6;                       // tiles of 64 columns
  int wave = blockIdx.x * 8 + (threadIdx.x >> 5);
  if (wave >= Mtiles * nT) return;        // wave-uniform
  int mt = wave / nT;
  int nt = wave - mt * nT;
  int m0 = mt << 4, n0 = nt << 6;
  int lane = threadIdx.x & 31;
  int r  = lane & 15;                     // row (A) / column (B,C) within tile
  int kq = lane >> 4;                     // half-select per ISA fragment layout

  // A 16x32 bf16 fragment: lanes 0-15 hold K[0..15]; 16-31 K[16..31]
  const __bf16* ap  = A  + (size_t)(m0 + r) * K + kq * 16;
  // B 32x16 fragment: lane = column, 16 contiguous K values (kq selects K half)
  const __bf16* bp0 = Bt + (size_t)(n0 + r) * K + kq * 16;
  const size_t bstep = (size_t)16 * K;

  v8f acc0 = {}, acc1 = {}, acc2 = {}, acc3 = {};
  for (int k0 = 0; k0 < K; k0 += 32) {
    union U { v16bf v; v8bf h[2]; } a, b0, b1, b2, b3;
    a.h[0] = *(const v8bf*)(ap + k0);
    a.h[1] = *(const v8bf*)(ap + k0 + 8);
    const __bf16* bp = bp0 + k0;
    b0.h[0] = *(const v8bf*)(bp);  b0.h[1] = *(const v8bf*)(bp + 8);  bp += bstep;
    b1.h[0] = *(const v8bf*)(bp);  b1.h[1] = *(const v8bf*)(bp + 8);  bp += bstep;
    b2.h[0] = *(const v8bf*)(bp);  b2.h[1] = *(const v8bf*)(bp + 8);  bp += bstep;
    b3.h[0] = *(const v8bf*)(bp);  b3.h[1] = *(const v8bf*)(bp + 8);
    acc0 = __builtin_amdgcn_wmma_f32_16x16x32_bf16(false, a.v, false, b0.v, (short)0, acc0, false, false);
    acc1 = __builtin_amdgcn_wmma_f32_16x16x32_bf16(false, a.v, false, b1.v, (short)0, acc1, false, false);
    acc2 = __builtin_amdgcn_wmma_f32_16x16x32_bf16(false, a.v, false, b2.v, (short)0, acc2, false, false);
    acc3 = __builtin_amdgcn_wmma_f32_16x16x32_bf16(false, a.v, false, b3.v, (short)0, acc3, false, false);
  }

  // C layout: lane 0-15 -> N=lane, VGPR i -> M=i; lanes 16-31 -> M=8+i
  v8f accs[4] = {acc0, acc1, acc2, acc3};
#pragma unroll
  for (int j = 0; j < 4; j++) {
    int col = n0 + j * 16 + r;
    float bv = bias ? bias[col] : 0.f;
#pragma unroll
    for (int i = 0; i < 8; i++) {
      int row = m0 + kq * 8 + i;
      if (row < Mvalid) C[(size_t)row * ldc + col] = accs[j][i] + bv;
    }
  }
}

// ---------------- attention logits per (node, head): wave reduction ----------------
__global__ void __launch_bounds__(256) node_alpha(
    const float* __restrict__ h, const float* __restrict__ asrc,
    const float* __restrict__ adst, float* __restrict__ als,
    float* __restrict__ ald, int NH) {
  int wv = blockIdx.x * 8 + (threadIdx.x >> 5);
  if (wv >= NH) return;
  int n  = wv >> 2;
  int hh = wv & 3;
  int lane = threadIdx.x & 31;
  const float* hp = h + (size_t)n * HC + hh * CH;
  const float* as = asrc + hh * CH;
  const float* ad = adst + hh * CH;
  float ss = 0.f, sd = 0.f;
#pragma unroll
  for (int j = lane; j < CH; j += 32) {
    float v = hp[j];
    ss += v * as[j];
    sd += v * ad[j];
  }
  for (int o = 16; o; o >>= 1) {
    ss += __shfl_xor(ss, o, 32);
    sd += __shfl_xor(sd, o, 32);
  }
  if (lane == 0) { als[wv] = ss; ald[wv] = sd; }
}

// ---------------- edge passes ----------------
__device__ __forceinline__ unsigned flipf(float f) {
  unsigned u = __float_as_uint(f);
  return (u & 0x80000000u) ? ~u : (u | 0x80000000u);
}
__device__ __forceinline__ float unflipf(unsigned u) {
  return __uint_as_float((u & 0x80000000u) ? (u ^ 0x80000000u) : ~u);
}

__global__ void edge_max(const int* __restrict__ ei, int E, int Etot,
                         const float* __restrict__ als, const float* __restrict__ ald,
                         unsigned* __restrict__ emax) {
  int e = blockIdx.x * blockDim.x + threadIdx.x;
  if (e >= Etot) return;
  int s = (e < E) ? ei[e]     : (e - E);
  int d = (e < E) ? ei[E + e] : (e - E);
#pragma unroll
  for (int hh = 0; hh < HD; hh++) {
    float v = als[s * HD + hh] + ald[d * HD + hh];
    v = (v > 0.f) ? v : NEG_SLOPE * v;
    atomicMax(&emax[d * HD + hh], flipf(v));
  }
}

__global__ void edge_exp(const int* __restrict__ ei, int E, int Etot,
                         const float* __restrict__ als, const float* __restrict__ ald,
                         const unsigned* __restrict__ emax,
                         float* __restrict__ wbuf, float* __restrict__ denom) {
  int e = blockIdx.x * blockDim.x + threadIdx.x;
  if (e >= Etot) return;
  int s = (e < E) ? ei[e]     : (e - E);
  int d = (e < E) ? ei[E + e] : (e - E);
#pragma unroll
  for (int hh = 0; hh < HD; hh++) {
    float v = als[s * HD + hh] + ald[d * HD + hh];
    v = (v > 0.f) ? v : NEG_SLOPE * v;
    float w = __expf(v - unflipf(emax[d * HD + hh]));
    wbuf[(size_t)e * HD + hh] = w;
    atomicAdd(&denom[d * HD + hh], w);
  }
}

// wave per edge: agg[dst] += alpha * h[src]
__global__ void __launch_bounds__(256) edge_agg(
    const int* __restrict__ ei, int E, int Etot,
    const float* __restrict__ h, const float* __restrict__ wbuf,
    const float* __restrict__ denom, float* __restrict__ agg) {
  int wv = blockIdx.x * 8 + (threadIdx.x >> 5);
  if (wv >= Etot) return;
  int s = (wv < E) ? ei[wv]     : (wv - E);
  int d = (wv < E) ? ei[E + wv] : (wv - E);
  int lane = threadIdx.x & 31;
  float alpha[HD];
#pragma unroll
  for (int hh = 0; hh < HD; hh++)
    alpha[hh] = wbuf[(size_t)wv * HD + hh] / denom[d * HD + hh];
  const float* hs = h   + (size_t)s * HC;
  float*       ag = agg + (size_t)d * HC;
  __builtin_prefetch((const void*)(hs + (lane << 5)), 0, 0);  // global_prefetch
#pragma unroll 4
  for (int j = lane; j < HC; j += 32) {
    atomicAdd(&ag[j], alpha[j >> 8] * hs[j]);
  }
}

// agg + bias -> relu -> bf16 (feeds next WMMA GEMM)
__global__ void post_bias_relu(const float* __restrict__ agg, const float* __restrict__ bias,
                               __bf16* __restrict__ hb, long long total) {
  long long i = (long long)blockIdx.x * blockDim.x + threadIdx.x;
  if (i >= total) return;
  int j = (int)(i & (HC - 1));
  float v = agg[i] + bias[j];
  hb[i] = (__bf16)(v > 0.f ? v : 0.f);
}

// ---------------- host launcher ----------------

extern "C" void kernel_launch(void* const* d_in, const int* in_sizes, int n_in,
                              void* d_out, int out_size, void* d_ws, size_t ws_size,
                              hipStream_t stream) {
  const float* x   = (const float*)d_in[0];
  const int*   ei  = (const int*)d_in[1];
  const float* W1  = (const float*)d_in[2];
  const float* as1 = (const float*)d_in[3];
  const float* ad1 = (const float*)d_in[4];
  const float* b1  = (const float*)d_in[5];
  const float* W2  = (const float*)d_in[6];
  const float* as2 = (const float*)d_in[7];
  const float* ad2 = (const float*)d_in[8];
  const float* b2  = (const float*)d_in[9];
  const float* Wfc = (const float*)d_in[10];
  const float* bfc = (const float*)d_in[11];
  float* out = (float*)d_out;

  const int DIN = 512, DOUT = 512;
  const int Nv   = in_sizes[0] / DIN;   // 10000 nodes
  const int E    = in_sizes[1] / 2;     // 160000 edges
  const int Etot = E + Nv;              // + self loops
  const int Mp   = (Nv + 15) & ~15;     // padded rows for 16-row WMMA tiles

  char* ws = (char*)d_ws;
  size_t off = 0;
  auto carve = [&](size_t bytes) -> char* {
    char* p = ws + off;
    off = (off + bytes + 255) & ~(size_t)255;
    return p;
  };
  __bf16*   xb    = (__bf16*)carve((size_t)Mp * DIN * 2);
  __bf16*   w1t   = (__bf16*)carve((size_t)HC * DIN * 2);
  __bf16*   w2t   = (__bf16*)carve((size_t)HC * HC * 2);
  __bf16*   wft   = (__bf16*)carve((size_t)DOUT * HC * 2);
  float*    h     = (float*)carve((size_t)Mp * HC * 4);
  float*    agg   = (float*)carve((size_t)Mp * HC * 4);
  __bf16*   hb    = (__bf16*)carve((size_t)Mp * HC * 2);
  float*    als   = (float*)carve((size_t)Nv * HD * 4);
  float*    ald   = (float*)carve((size_t)Nv * HD * 4);
  unsigned* emax  = (unsigned*)carve((size_t)Nv * HD * 4);
  float*    denom = (float*)carve((size_t)Nv * HD * 4);
  float*    wbuf  = (float*)carve((size_t)Etot * HD * 4);
  (void)ws_size; (void)n_in; (void)out_size;

  // pack inputs / weights to bf16 (weights transposed for contiguous B fragments)
  { long long n = (long long)Mp * DIN;
    cvt_pad<<<(unsigned)((n + 255) / 256), 256, 0, stream>>>(x, xb, Nv, DIN, n); }
  { long long n = (long long)DIN * HC;
    cvt_transpose<<<(unsigned)((n + 255) / 256), 256, 0, stream>>>(W1, w1t, DIN, HC); }
  { long long n = (long long)HC * HC;
    cvt_transpose<<<(unsigned)((n + 255) / 256), 256, 0, stream>>>(W2, w2t, HC, HC); }
  { long long n = (long long)HC * DOUT;
    cvt_transpose<<<(unsigned)((n + 255) / 256), 256, 0, stream>>>(Wfc, wft, HC, DOUT); }

  const int Mtiles = Mp / 16;
  auto gemm = [&](const __bf16* A, const __bf16* Bt, float* C, const float* bias,
                  int K, int Nw, int Mvalid) {
    int waves = Mtiles * (Nw >> 6);
    gemm_wmma<<<(unsigned)((waves + 7) / 8), 256, 0, stream>>>(A, Bt, C, bias, K, Nw,
                                                               Mtiles, Mvalid, Nw);
  };

  auto layer = [&](const __bf16* Fin, int Kin, const __bf16* Wt,
                   const float* asr, const float* adt, const float* bias) {
    gemm(Fin, Wt, h, nullptr, Kin, HC, Mp);
    { int NH = Nv * HD;
      node_alpha<<<(unsigned)((NH + 7) / 8), 256, 0, stream>>>(h, asr, adt, als, ald, NH); }
    { long long n = (long long)Nv * HD;   // flip(-inf) = 0x007FFFFF
      fill_u32<<<(unsigned)((n + 255) / 256), 256, 0, stream>>>(emax, 0x007FFFFFu, n); }
    { long long n = (long long)Nv * HD;
      fill_u32<<<(unsigned)((n + 255) / 256), 256, 0, stream>>>((unsigned*)denom, 0u, n); }
    { long long n = (long long)Mp * HC;
      fill_u32<<<(unsigned)((n + 255) / 256), 256, 0, stream>>>((unsigned*)agg, 0u, n); }
    edge_max<<<(unsigned)((Etot + 255) / 256), 256, 0, stream>>>(ei, E, Etot, als, ald, emax);
    edge_exp<<<(unsigned)((Etot + 255) / 256), 256, 0, stream>>>(ei, E, Etot, als, ald,
                                                                 emax, wbuf, denom);
    edge_agg<<<(unsigned)((Etot + 7) / 8), 256, 0, stream>>>(ei, E, Etot, h, wbuf, denom, agg);
    { long long n = (long long)Mp * HC;
      post_bias_relu<<<(unsigned)((n + 255) / 256), 256, 0, stream>>>(agg, bias, hb, n); }
  };

  layer(xb, DIN, w1t, as1, ad1, b1);   // GAT layer 1
  layer(hb, HC,  w2t, as2, ad2, b2);   // GAT layer 2 (reads hb, rewrites hb at end)
  gemm(hb, wft, out, bfc, HC, DOUT, Nv);  // FC head, bias fused, rows guarded
}
